// ParallelComputing_15702400434467
// MI455X (gfx1250) — matrix operand order repaired and checked by reading, not compiled, not run on hardware
//
#include <hip/hip_runtime.h>
#include <hip/hip_bf16.h>
#include <cstdint>

#define N_ROWS 8192
#define D_DIS  4096
#define IN_CH  512
#define OUT_CH 256
#define X_DIM  512

typedef __attribute__((ext_vector_type(16))) __bf16 v16bf;
typedef __attribute__((ext_vector_type(8)))  float  v8f;

struct BPack { uint4 lo; uint4 hi; };
union  BCast { BPack p; v16bf v; };

// ---- CDNA5 async memory->LDS path (ASYNCcnt), with synchronous fallback ----
#if defined(__gfx1250__) && \
    __has_builtin(__builtin_amdgcn_global_load_async_to_lds_b128) && \
    __has_builtin(__builtin_amdgcn_s_wait_asynccnt)
#define HAVE_ASYNC_LDS 1
#else
#define HAVE_ASYNC_LDS 0
#endif

#if HAVE_ASYNC_LDS
typedef int v4i_async __attribute__((vector_size(16)));
typedef __attribute__((address_space(1))) v4i_async* async_gptr;
typedef __attribute__((address_space(3))) v4i_async* async_lptr;
#endif

__device__ __forceinline__ void copy16_g2s(__bf16* ldst, const __bf16* gsrc) {
#if HAVE_ASYNC_LDS
  __builtin_amdgcn_global_load_async_to_lds_b128(
      (async_gptr)gsrc, (async_lptr)ldst, 0, 0);
#else
  *reinterpret_cast<uint4*>(ldst) = *reinterpret_cast<const uint4*>(gsrc);
#endif
}

__device__ __forceinline__ void async_join() {
#if HAVE_ASYNC_LDS
  __builtin_amdgcn_s_wait_asynccnt(0);
#endif
}

// A-operand (16x32 bf16, M x K): lane = M row; elems 0-7 = K[8*hi .. 8*hi+7],
// elems 8-15 = K[16+8*hi .. 23+8*hi]   (ISA 7.12.2 16-bit A layout)
__device__ __forceinline__ v16bf load_a_bf16(const __bf16* base, int ld, int row,
                                             int hi, int koff) {
  const __bf16* p = base + (size_t)row * ld + koff + (hi << 3);
  BCast bc;
  bc.p.lo = *reinterpret_cast<const uint4*>(p);
  bc.p.hi = *reinterpret_cast<const uint4*>(p + 16);
  return bc.v;
}

// B-operand (32x16 bf16, K x N) read from Bt (row-major, rows are B columns):
// lane n holds K[8*hi .. 8*hi+7] and K[16+8*hi .. 23+8*hi] of column n
// (same k order as the A operand)
__device__ __forceinline__ v16bf load_b_bf16(const __bf16* baseT, int ld, int n,
                                             int hi, int koff) {
  const __bf16* p = baseT + (size_t)n * ld + koff + (hi << 3);
  BCast bc;
  bc.p.lo = *reinterpret_cast<const uint4*>(p);
  bc.p.hi = *reinterpret_cast<const uint4*>(p + 16);
  return bc.v;
}

__device__ __forceinline__ v8f wmma_bf16(v16bf a, v16bf b, v8f c) {
  return __builtin_amdgcn_wmma_f32_16x16x32_bf16(false, a, false, b, (short)0, c,
                                                 false, false);
}

// ---------------------------------------------------------------- prep kernels

__global__ __launch_bounds__(256) void cvt_f32_to_bf16(const float* __restrict__ s,
                                                       __bf16* __restrict__ d, int n) {
  int i = blockIdx.x * blockDim.x + threadIdx.x;
  if (i < n) d[i] = (__bf16)s[i];
}

// d[c*R + r] = bf16(s[r*C + c])   (build transposed bf16 copy)
__global__ __launch_bounds__(256) void cvt_t_f32_to_bf16(const float* __restrict__ s,
                                                         __bf16* __restrict__ d,
                                                         int R, int C) {
  int i = blockIdx.x * blockDim.x + threadIdx.x;
  if (i < R * C) {
    int r = i / C, c = i - r * C;
    d[(size_t)c * R + r] = (__bf16)s[i];
  }
}

// tiled bf16 transpose: dst[C][R] = src[R][C]^T
__global__ __launch_bounds__(256) void transpose_bf16(const __bf16* __restrict__ src,
                                                      __bf16* __restrict__ dst,
                                                      int R, int C) {
  __shared__ __bf16 t[32][33];
  int c0 = blockIdx.x * 32, r0 = blockIdx.y * 32;
#pragma unroll
  for (int i = 0; i < 4; ++i) {
    int r = threadIdx.y + i * 8;
    t[r][threadIdx.x] = src[(size_t)(r0 + r) * C + c0 + threadIdx.x];
  }
  __syncthreads();
#pragma unroll
  for (int i = 0; i < 4; ++i) {
    int r = threadIdx.y + i * 8;
    dst[(size_t)(c0 + r) * R + r0 + threadIdx.x] = t[threadIdx.x][r];
  }
}

// C[M,Nc] (bf16) = A[M,K](bf16) @ Bt[Nc,K](bf16)^T ; one 16x16 tile per wave
__global__ __launch_bounds__(128) void gemm_bf16_tn(const __bf16* __restrict__ A,
                                                    const __bf16* __restrict__ Bt,
                                                    __bf16* __restrict__ Cout,
                                                    int M, int Nc, int K) {
  const int lane = threadIdx.x & 31;
  const int r16  = lane & 15;
  const int hi   = lane >> 4;
  const int w    = threadIdx.x >> 5;
  int tile = blockIdx.x * 4 + w;
  int ntn  = Nc >> 4;
  int mt = tile / ntn, nt = tile - mt * ntn;
  if (mt * 16 >= M) return;  // wave-uniform
  const __bf16* Ab = A  + (size_t)mt * 16 * K;
  const __bf16* Bb = Bt + (size_t)nt * 16 * K;
  v8f acc = {};
  for (int kk = 0; kk < K; kk += 32) {
    v16bf a = load_a_bf16(Ab, K, r16, hi, kk);
    v16bf b = load_b_bf16(Bb, K, r16, hi, kk);
    acc = wmma_bf16(a, b, acc);
  }
#pragma unroll
  for (int r = 0; r < 8; ++r)
    Cout[(size_t)(mt * 16 + r + 8 * hi) * Nc + nt * 16 + r16] = (__bf16)acc[r];
}

// ------------------------------------------------------------- fused main pass
// Per wave: 16 rows. scores chunk -> mask -> contrib accumulate, then f1 + f2.
// LDS (160KB dynamic): Xs[64][512] | RA(32K) | RB(32K) | RC(32K)
//   main phase: RA=HbS[32][512], RB=HtS[512][32], RC=per-wave S buffers
//   f2 phase:   RA..RB = Cs[64][512] (contrib bf16), RC = WrtS[512][32]
__global__ __launch_bounds__(128, 1) void fused_attention_kernel(
    const float*  __restrict__ xf,    // [N, 512] f32
    const int*    __restrict__ mat,   // [N, 4096] int32
    const __bf16* __restrict__ Xb,    // [N, 512] bf16
    const __bf16* __restrict__ Hb,    // [4096, 512] bf16
    const __bf16* __restrict__ Hbt,   // [512, 4096] bf16
    const __bf16* __restrict__ Wrtb,  // [512, 1024] bf16 (dim_reduction_mat^T)
    float* __restrict__ out)          // [N, 1024] f32
{
  extern __shared__ __align__(16) char smem[];
  __bf16* Xs = reinterpret_cast<__bf16*>(smem);  // [64][512]
  __bf16* RA = Xs + 64 * 512;
  __bf16* RB = RA + 32 * 512;
  __bf16* RC = RB + 512 * 32;

  const int tid  = threadIdx.x;
  const int lane = tid & 31;
  const int w    = tid >> 5;
  const int r16  = lane & 15;
  const int hi   = lane >> 4;
  const int bm0  = blockIdx.x * 64;
  const int m0w  = bm0 + w * 16;

  // stage the block's 64-row X strip (bf16) once (async: joined before first use)
#pragma unroll
  for (int i = 0; i < 32; ++i) {
    int idx = tid + i * 128;
    int row = idx >> 6;
    int c8  = (idx & 63) * 8;
    copy16_g2s(Xs + row * 512 + c8, Xb + (size_t)(bm0 + row) * 512 + c8);
  }

  v8f acc[32];
#pragma unroll
  for (int ct = 0; ct < 32; ++ct) acc[ct] = {};

  __bf16*       HbS = RA;                   // [32][512]
  __bf16*       HtS = RB;                   // [512][32]
  __bf16*       Ssw = RC + w * (16 * 32);   // [16][32] per wave
  const __bf16* Xsw = Xs + w * 16 * 512;

  for (int dc = 0; dc < D_DIS / 32; ++dc) {
    __syncthreads();
    // stage Hb chunk [32][512] and Hbt chunk [512][32] via async LDS DMA
#pragma unroll
    for (int i = 0; i < 16; ++i) {
      int idx = tid + i * 128;
      int row = idx >> 6;
      int c8  = (idx & 63) * 8;
      copy16_g2s(HbS + row * 512 + c8,
                 Hb + (size_t)(dc * 32 + row) * 512 + c8);
    }
#pragma unroll
    for (int i = 0; i < 16; ++i) {
      int idx = tid + i * 128;
      int x   = idx >> 2;
      int q8  = (idx & 3) * 8;
      copy16_g2s(HtS + x * 32 + q8,
                 Hbt + (size_t)x * D_DIS + dc * 32 + q8);
    }
    async_join();
    __syncthreads();

    // stage 1: S[16 rows][32 diseases] = X . H_chunk^T, masked, to bf16 in LDS
#pragma unroll
    for (int nt = 0; nt < 2; ++nt) {
      v8f s = {};
#pragma unroll
      for (int kc = 0; kc < 16; ++kc) {
        v16bf a = load_a_bf16(Xsw, 512, r16, hi, kc * 32);
        v16bf b = load_b_bf16(HbS + nt * 16 * 512, 512, r16, hi, kc * 32);
        s = wmma_bf16(a, b, s);
      }
      int dcol = dc * 32 + nt * 16 + r16;
#pragma unroll
      for (int r = 0; r < 8; ++r) {
        int row = m0w + r + 8 * hi;
        int mk  = __builtin_nontemporal_load(mat + (size_t)row * D_DIS + dcol);
        float v = (mk != 0) ? s[r] : 0.0f;
        Ssw[(r + 8 * hi) * 32 + nt * 16 + r16] = (__bf16)v;
      }
    }

    // stage 2: contrib[16][512] += S . H_chunk   (B from transposed chunk)
    v16bf a2 = load_a_bf16(Ssw, 32, r16, hi, 0);
#pragma unroll
    for (int ct = 0; ct < 32; ++ct) {
      v16bf b2 = load_b_bf16(HtS + ct * 16 * 32, 32, r16, hi, 0);
      acc[ct] = wmma_bf16(a2, b2, acc[ct]);
    }
  }

  __syncthreads();
  // f1 = x + contrib -> out[:, :512]; contrib bf16 copy into LDS for f2
  __bf16* Cs  = RA;  // [64][512]
  __bf16* Csw = Cs + w * 16 * 512;
#pragma unroll
  for (int ct = 0; ct < 32; ++ct) {
#pragma unroll
    for (int r = 0; r < 8; ++r) {
      int row = m0w + r + 8 * hi;
      int col = ct * 16 + r16;
      float c = acc[ct][r];
      float xv = __builtin_nontemporal_load(xf + (size_t)row * 512 + col);
      out[(size_t)row * 1024 + col] = xv + c;
      Csw[(r + 8 * hi) * 512 + col] = (__bf16)c;
    }
  }

  // f2 = concat(x, contrib) @ Wr -> out[:, 512:]
#pragma unroll
  for (int ct = 0; ct < 32; ++ct) acc[ct] = {};
  __bf16* WrtS = RC;  // [512][32]
  for (int kc = 0; kc < 32; ++kc) {
    __syncthreads();
#pragma unroll
    for (int i = 0; i < 16; ++i) {
      int idx = tid + i * 128;
      int x   = idx >> 2;
      int q8  = (idx & 3) * 8;
      copy16_g2s(WrtS + x * 32 + q8,
                 Wrtb + (size_t)x * 1024 + kc * 32 + q8);
    }
    async_join();
    __syncthreads();
    int kk = kc * 32;
    const __bf16* abase = (kk < 512) ? Xsw : Csw;  // uniform branch
    v16bf a = load_a_bf16(abase, 512, r16, hi, kk & 511);
#pragma unroll
    for (int ct = 0; ct < 32; ++ct) {
      v16bf b = load_b_bf16(WrtS + ct * 16 * 32, 32, r16, hi, 0);
      acc[ct] = wmma_bf16(a, b, acc[ct]);
    }
  }
#pragma unroll
  for (int ct = 0; ct < 32; ++ct) {
#pragma unroll
    for (int r = 0; r < 8; ++r) {
      int row = m0w + r + 8 * hi;
      out[(size_t)row * 1024 + 512 + ct * 16 + r16] = acc[ct][r];
    }
  }
}

// --------------------------------------------------------------------- launch

extern "C" void kernel_launch(void* const* d_in, const int* in_sizes, int n_in,
                              void* d_out, int out_size, void* d_ws, size_t ws_size,
                              hipStream_t stream) {
  (void)in_sizes; (void)n_in; (void)out_size; (void)ws_size;
  const float* xf = (const float*)d_in[0];   // [8192, 512]
  const float* df = (const float*)d_in[1];   // [4096, 512]
  const float* w1 = (const float*)d_in[2];   // [512, 256]
  const float* w2 = (const float*)d_in[3];   // [256, 512]
  const float* wr = (const float*)d_in[4];   // [1024, 512]
  const int*  mat = (const int*)d_in[5];     // [8192, 4096]
  float* out = (float*)d_out;

  char* ws = (char*)d_ws;
  size_t off = 0;
  auto carve = [&](size_t bytes) -> char* {
    char* p = ws + off;
    off += (bytes + 255) & ~(size_t)255;
    return p;
  };
  __bf16* Xb   = (__bf16*)carve((size_t)N_ROWS * X_DIM * 2);   // 8 MB
  __bf16* DFb  = (__bf16*)carve((size_t)D_DIS * IN_CH * 2);    // 4 MB
  __bf16* Tb   = (__bf16*)carve((size_t)D_DIS * OUT_CH * 2);   // 2 MB
  __bf16* W1tb = (__bf16*)carve((size_t)OUT_CH * IN_CH * 2);   // 256 KB
  __bf16* W2tb = (__bf16*)carve((size_t)X_DIM * OUT_CH * 2);   // 256 KB
  __bf16* Wrtb = (__bf16*)carve((size_t)X_DIM * 2 * X_DIM * 2);// 1 MB
  __bf16* Hb   = (__bf16*)carve((size_t)D_DIS * X_DIM * 2);    // 4 MB
  __bf16* Hbt  = (__bf16*)carve((size_t)X_DIM * D_DIS * 2);    // 4 MB

  // bf16 copies + transposed weight copies
  {
    int n = N_ROWS * X_DIM;
    cvt_f32_to_bf16<<<(n + 255) / 256, 256, 0, stream>>>(xf, Xb, n);
  }
  {
    int n = D_DIS * IN_CH;
    cvt_f32_to_bf16<<<(n + 255) / 256, 256, 0, stream>>>(df, DFb, n);
  }
  cvt_t_f32_to_bf16<<<(IN_CH * OUT_CH + 255) / 256, 256, 0, stream>>>(w1, W1tb, IN_CH, OUT_CH);
  cvt_t_f32_to_bf16<<<(OUT_CH * X_DIM + 255) / 256, 256, 0, stream>>>(w2, W2tb, OUT_CH, X_DIM);
  cvt_t_f32_to_bf16<<<(2 * X_DIM * X_DIM + 255) / 256, 256, 0, stream>>>(wr, Wrtb, 2 * X_DIM, X_DIM);

  // T = disease_feature @ dis_trans   [4096, 256]
  gemm_bf16_tn<<<(D_DIS / 16) * (OUT_CH / 16) / 4, 128, 0, stream>>>(
      DFb, W1tb, Tb, D_DIS, OUT_CH, IN_CH);
  // H = T @ disease_dimension_transform   [4096, 512]
  gemm_bf16_tn<<<(D_DIS / 16) * (X_DIM / 16) / 4, 128, 0, stream>>>(
      Tb, W2tb, Hb, D_DIS, X_DIM, OUT_CH);
  // H^T  [512, 4096]
  transpose_bf16<<<dim3(X_DIM / 32, D_DIS / 32), dim3(32, 8), 0, stream>>>(
      Hb, Hbt, D_DIS, X_DIM);

  // fused scores+mask+contrib+f1+f2
  const size_t lds_bytes = (64 * 512 + 3 * 32 * 512) * sizeof(__bf16);  // 160 KB
  fused_attention_kernel<<<N_ROWS / 64, 128, lds_bytes, stream>>>(
      xf, mat, Xb, Hb, Hbt, Wrtb, out);
}
